// SNN1_85925115724560
// MI455X (gfx1250) — matrix operand order repaired and checked by reading, not compiled, not run on hardware
//
#include <hip/hip_runtime.h>

typedef __attribute__((ext_vector_type(2)))  float    v2f;
typedef __attribute__((ext_vector_type(8)))  float    v8f;
typedef __attribute__((ext_vector_type(8)))  _Float16 v8h;
typedef __attribute__((ext_vector_type(16))) _Float16 v16h;

#define BATCH   128
#define TSTEPS  100
#define FDIM    2048
#define BETA    0.9f
#define THRESH  1.0f

// ---- persistent f16-split path: 128 blocks x 8 waves = 1024 waves = 8 Mtiles x 128 Ntiles
#define NBLK 128
#define WPB  8

// ---- workspace layout (bytes) ----
#define WS_CNT_OFF  ((size_t)0)
#define WS_WHI_OFF  ((size_t)256)
#define WS_WLO_OFF  (WS_WHI_OFF + (size_t)FDIM * FDIM * 2)
#define WS_SPK0_OFF (WS_WLO_OFF + (size_t)FDIM * FDIM * 2)
#define WS_SPK1_OFF (WS_SPK0_OFF + (size_t)BATCH * FDIM * 2)
#define WS_NEED     (WS_SPK1_OFF + (size_t)BATCH * FDIM * 2)

__global__ void snn_zero_counter(unsigned* c) { *c = 0u; }

// Split f32 W into Whi + Wlo (f16 each): Whi = f16(w), Wlo = f16(w - f32(Whi)).
// spk ∈ {0,1} is exact in f16, so spk@Whi^T + spk@Wlo^T ≈ f32 GEMM (~2^-22 rel err).
__global__ void snn_split_w(const float* __restrict__ W,
                            _Float16* __restrict__ whi, _Float16* __restrict__ wlo) {
  const int n = FDIM * FDIM;
  for (int i = blockIdx.x * blockDim.x + threadIdx.x; i < n; i += gridDim.x * blockDim.x) {
    const float w = W[i];
    const _Float16 hi = (_Float16)w;
    whi[i] = hi;
    wlo[i] = (_Float16)(w - (float)hi);
  }
}

// Persistent SNN: each wave owns a fixed 16(M)x16(N) tile for all timesteps.
// mem state lives in registers; spikes ping-pong through f16 workspace buffers;
// the block's t-invariant B tile (Whi/Wlo rows) is staged ONCE in 128 KB of LDS
// (gfx1250: 320 KB per WGP), hi/lo interleaved per row so all DS reads use
// 16-bit immediate offsets off a single base.
__global__ __launch_bounds__(256) void snn_persistent_f16(
    const float* __restrict__ x,        // [B,T,F]
    const float* __restrict__ bias,     // [F]
    float* __restrict__ spk_rec,        // [B,T,F]
    float* __restrict__ mem_rec,        // [B,T,F]
    const _Float16* __restrict__ whi,   // [F,F] f16 high part of W
    const _Float16* __restrict__ wlo,   // [F,F] f16 low  part of W
    _Float16* __restrict__ spk16a,      // [B,F] ping
    _Float16* __restrict__ spk16b,      // [B,F] pong
    unsigned* __restrict__ bar)
{
  __shared__ _Float16 ldsb[16 * 2 * FDIM];   // 128 KB: [row][ hi FDIM | lo FDIM ]

  const int lane = threadIdx.x & 31;
  const int wave = threadIdx.x >> 5;
  const int half = lane >> 4;
  const int lrow = lane & 15;

  const int mbase = wave << 4;        // 8 M tiles of 16 (wave id)
  const int nbase = blockIdx.x << 4;  // 128 N tiles of 16 (block id)
  const int n     = nbase + lrow;     // this lane's output column

  // ---- stage the block's B tile (t-invariant) into LDS, hi/lo interleaved per row
  {
    const _Float16* ghi = whi + (size_t)nbase * FDIM;
    const _Float16* glo = wlo + (size_t)nbase * FDIM;
    for (int i = threadIdx.x * 8; i < 16 * FDIM; i += 256 * 8) {
      const int row = i >> 11;           // i / FDIM
      const int k   = i & (FDIM - 1);    // i % FDIM
      *(v8h*)&ldsb[row * 2 * FDIM + k]        = *(const v8h*)(ghi + i);
      *(v8h*)&ldsb[row * 2 * FDIM + FDIM + k] = *(const v8h*)(glo + i);
    }
  }
  __syncthreads();

  const size_t rowTF = (size_t)TSTEPS * FDIM;
  const float  bn    = bias[n];

  // B fragment base in LDS: lane holds column n (local row lrow), K = kk + 16*half .. +15.
  // hi at [bofs + kk], lo at [bofs + FDIM + kk]  (all within 16-bit DS offsets).
  const int bofs = lrow * 2 * FDIM + 16 * half;
  // A fragment row base within spk16 buffer: lane M = mbase+lrow,
  // K = kk + 16*half + {0..15}.
  const size_t arow_off = (size_t)(mbase + lrow) * FDIM + 16 * half;
  // Output element (m0, n) flat index at t=0, with m_r = m0 + r.
  const size_t b0 = (size_t)(mbase + 8 * half) * rowTF + (size_t)n;

  v8f memst = {};   // register-resident membrane state (one column, 8 rows)

  for (int t = 0; t < TSTEPS; ++t) {
    v8f acc = {};
    if (t > 0) {
      const _Float16* ar = (((t - 1) & 1) ? spk16b : spk16a) + arow_off;
      // 4 independent accumulators (hi/lo x even/odd K-step) -> WMMA RAW
      // dependency distance 4, keeps the matrix pipe full.
      v8f ah0 = {}, al0 = {}, ah1 = {}, al1 = {};
      for (int kk = 0; kk < FDIM; kk += 64) {
        union { v16h v; v8h h[2]; } a0, a1, bh0, bl0, bh1, bl1;
        a0.h[0]  = *(const v8h*)(ar + kk);
        a0.h[1]  = *(const v8h*)(ar + kk + 8);
        a1.h[0]  = *(const v8h*)(ar + kk + 32);
        a1.h[1]  = *(const v8h*)(ar + kk + 40);
        bh0.h[0] = *(const v8h*)&ldsb[bofs + kk];
        bh0.h[1] = *(const v8h*)&ldsb[bofs + kk + 8];
        bl0.h[0] = *(const v8h*)&ldsb[bofs + FDIM + kk];
        bl0.h[1] = *(const v8h*)&ldsb[bofs + FDIM + kk + 8];
        bh1.h[0] = *(const v8h*)&ldsb[bofs + kk + 32];
        bh1.h[1] = *(const v8h*)&ldsb[bofs + kk + 40];
        bl1.h[0] = *(const v8h*)&ldsb[bofs + FDIM + kk + 32];
        bl1.h[1] = *(const v8h*)&ldsb[bofs + FDIM + kk + 40];
        ah0 = __builtin_amdgcn_wmma_f32_16x16x32_f16(false, a0.v, false, bh0.v, (short)0, ah0, false, false);
        al0 = __builtin_amdgcn_wmma_f32_16x16x32_f16(false, a0.v, false, bl0.v, (short)0, al0, false, false);
        ah1 = __builtin_amdgcn_wmma_f32_16x16x32_f16(false, a1.v, false, bh1.v, (short)0, ah1, false, false);
        al1 = __builtin_amdgcn_wmma_f32_16x16x32_f16(false, a1.v, false, bl1.v, (short)0, al1, false, false);
      }
      acc = (ah0 + ah1) + (al0 + al1);
    }

    _Float16* scur = ((t & 1) ? spk16b : spk16a);
    const size_t off = b0 + (size_t)t * FDIM;
#pragma unroll
    for (int r = 0; r < 8; ++r) {
      const float mprev = memst[r];
      const float reset = (mprev > THRESH) ? THRESH : 0.0f;
      const float mem   = BETA * mprev + x[off + (size_t)r * rowTF] + acc[r] + bn - reset;
      memst[r] = mem;
      const float s = (mem > THRESH) ? 1.0f : 0.0f;
      mem_rec[off + (size_t)r * rowTF] = mem;
      spk_rec[off + (size_t)r * rowTF] = s;
      scur[(size_t)(mbase + 8 * half + r) * FDIM + n] = (_Float16)s;
    }

    // Prefetch next timestep's x slice; overlaps with the grid barrier.
    if (t + 1 < TSTEPS) {
#pragma unroll
      for (int r = 0; r < 8; ++r)
        __builtin_prefetch(x + off + FDIM + (size_t)r * rowTF, 0, 0);
    }

    // Grid barrier: monotonic counter, one arrival per block.
    __syncthreads();
    if (threadIdx.x == 0) {
      __threadfence();   // publish this block's spk16 stores
      atomicAdd(bar, 1u);
      const unsigned target = (unsigned)NBLK * (unsigned)(t + 1);
      while (__hip_atomic_load(bar, __ATOMIC_RELAXED, __HIP_MEMORY_SCOPE_AGENT) < target) {
        __builtin_amdgcn_s_sleep(1);
      }
    }
    __syncthreads();
    __threadfence();     // acquire: drop stale near caches before reading others' spikes
  }
}

// ---------------- fallback: exact-f32 WMMA, one launch per timestep ----------------
__device__ __forceinline__ void snn_step_wave_f32(
    const float* __restrict__ x, const float* __restrict__ W,
    const float* __restrict__ bias, float* __restrict__ spk_rec,
    float* __restrict__ mem_rec, int t, int mbase, int nbase, int half, int lrow)
{
  const size_t rowTF = (size_t)TSTEPS * FDIM;
  v8f acc0 = {}; v8f acc1 = {}; v8f acc2 = {}; v8f acc3 = {};

  if (t > 0) {
    const float* arow = spk_rec + (size_t)(mbase + lrow) * rowTF + (size_t)(t - 1) * FDIM;
    const float* w0 = W + (size_t)(nbase +  0 + lrow) * FDIM;
    const float* w1 = W + (size_t)(nbase + 16 + lrow) * FDIM;
    const float* w2 = W + (size_t)(nbase + 32 + lrow) * FDIM;
    const float* w3 = W + (size_t)(nbase + 48 + lrow) * FDIM;
    const int kh = 2 * half;
#pragma unroll 2
    for (int kk = 0; kk < FDIM; kk += 4) {
      const int k0 = kk + kh;
      v2f a  = *(const v2f*)(arow + k0);
      v2f b0 = *(const v2f*)(w0 + k0);
      v2f b1 = *(const v2f*)(w1 + k0);
      v2f b2 = *(const v2f*)(w2 + k0);
      v2f b3 = *(const v2f*)(w3 + k0);
      acc0 = __builtin_amdgcn_wmma_f32_16x16x4_f32(false, a, false, b0, (short)0, acc0, false, false);
      acc1 = __builtin_amdgcn_wmma_f32_16x16x4_f32(false, a, false, b1, (short)0, acc1, false, false);
      acc2 = __builtin_amdgcn_wmma_f32_16x16x4_f32(false, a, false, b2, (short)0, acc2, false, false);
      acc3 = __builtin_amdgcn_wmma_f32_16x16x4_f32(false, a, false, b3, (short)0, acc3, false, false);
    }
  }

#define SNN_EPILOGUE(ACC, J) do {                                              \
    const int nn = nbase + (J) * 16 + lrow;                                    \
    const float bnn = bias[nn];                                                \
    _Pragma("unroll")                                                          \
    for (int r = 0; r < 8; ++r) {                                              \
      const int m = mbase + 8 * half + r;                                      \
      const size_t base = (size_t)m * rowTF + (size_t)t * FDIM + (size_t)nn;   \
      const float mprev = (t > 0) ? mem_rec[base - FDIM] : 0.0f;               \
      const float reset = (mprev > THRESH) ? THRESH : 0.0f;                    \
      const float mem   = BETA * mprev + x[base] + (ACC)[r] + bnn - reset;     \
      mem_rec[base] = mem;                                                     \
      spk_rec[base] = (mem > THRESH) ? 1.0f : 0.0f;                            \
    }                                                                          \
  } while (0)
  SNN_EPILOGUE(acc0, 0);
  SNN_EPILOGUE(acc1, 1);
  SNN_EPILOGUE(acc2, 2);
  SNN_EPILOGUE(acc3, 3);
#undef SNN_EPILOGUE
}

__global__ __launch_bounds__(256) void snn_step_kernel(
    const float* __restrict__ x, const float* __restrict__ W,
    const float* __restrict__ bias, float* __restrict__ spk_rec,
    float* __restrict__ mem_rec, int t)
{
  const int lane = threadIdx.x & 31;
  const int wave = threadIdx.x >> 5;
  const int half = lane >> 4;
  const int lrow = lane & 15;
  const int gw    = blockIdx.x * WPB + wave;  // 0..255
  const int mbase = (gw & 7) << 4;
  const int nbase = (gw >> 3) << 6;
  snn_step_wave_f32(x, W, bias, spk_rec, mem_rec, t, mbase, nbase, half, lrow);
}

extern "C" void kernel_launch(void* const* d_in, const int* in_sizes, int n_in,
                              void* d_out, int out_size, void* d_ws, size_t ws_size,
                              hipStream_t stream) {
  (void)in_sizes; (void)n_in; (void)out_size;
  const float* x    = (const float*)d_in[0];   // [128,100,2048] f32
  const float* W    = (const float*)d_in[1];   // [2048,2048] f32
  const float* bias = (const float*)d_in[2];   // [2048] f32
  float* spk_rec = (float*)d_out;
  float* mem_rec = spk_rec + (size_t)BATCH * TSTEPS * FDIM;

  if (ws_size >= WS_NEED) {
    char* ws = (char*)d_ws;
    unsigned*  bar    = (unsigned*)(ws + WS_CNT_OFF);
    _Float16*  whi    = (_Float16*)(ws + WS_WHI_OFF);
    _Float16*  wlo    = (_Float16*)(ws + WS_WLO_OFF);
    _Float16*  spk16a = (_Float16*)(ws + WS_SPK0_OFF);
    _Float16*  spk16b = (_Float16*)(ws + WS_SPK1_OFF);
    snn_split_w<<<dim3(1024), dim3(256), 0, stream>>>(W, whi, wlo);
    snn_zero_counter<<<dim3(1), dim3(1), 0, stream>>>(bar);
    snn_persistent_f16<<<dim3(NBLK), dim3(256), 0, stream>>>(
        x, bias, spk_rec, mem_rec, whi, wlo, spk16a, spk16b, bar);
  } else {
    for (int t = 0; t < TSTEPS; ++t) {
      snn_step_kernel<<<dim3(32), dim3(256), 0, stream>>>(x, W, bias, spk_rec, mem_rec, t);
    }
  }
}
